// XNORLinear_5781025980877
// MI455X (gfx1250) — 8-bit matrix operand order repaired and checked by reading, compiled, not run on hardware
//
#include <hip/hip_runtime.h>

#define BDIM 16384
#define IN   4096
#define OUT  4096
#define KT   (IN / 128)

typedef __attribute__((ext_vector_type(16))) int   v16i;
typedef __attribute__((ext_vector_type(8)))  float v8f;

static __device__ __forceinline__ unsigned sign_fp8(float v) {
    return v > 0.0f ? 0x38u : (v < 0.0f ? 0xB8u : 0x00u);
}

__global__ __launch_bounds__(256) void pack_x_kernel(const float* __restrict__ x,
                                                     unsigned* __restrict__ pa) {
    const int d      = blockIdx.x * 256 + threadIdx.x;
    const int tile   = d >> 9;
    const int within = d & 511;
    const int lane   = within >> 4;
    const int v      = within & 15;
    const int tm     = tile / KT;
    const int tk     = tile % KT;
    const int m      = tm * 16 + (lane & 15);
    const int h      = lane >> 4;
    const int vv     = v & 7;
    const int kb     = (v >> 2) * 32 + h * 16 + (v & 3) * 4;
    const int k      = tk * 128 + kb;

    const float4 f = *(const float4*)(x + (size_t)m * IN + k);
    const unsigned b0 = sign_fp8(f.x), b1 = sign_fp8(f.y);
    const unsigned b2 = sign_fp8(f.z), b3 = sign_fp8(f.w);
    pa[d] = b0 | (b1 << 8) | (b2 << 16) | (b3 << 24);
}

__global__ __launch_bounds__(256) void pack_w_kernel(const float* __restrict__ wfull,
                                                     unsigned char* __restrict__ pb,
                                                     float* __restrict__ scale) {
    const int o   = blockIdx.x;
    const int tid = threadIdx.x;
    const float* row = wfull + (size_t)o * IN;
    __shared__ float red[256];

    float s = 0.0f;
    for (int k = tid; k < IN; k += 256) s += row[k];
    red[tid] = s;
    __syncthreads();
    for (int off = 128; off > 0; off >>= 1) {
        if (tid < off) red[tid] += red[tid + off];
        __syncthreads();
    }
    const float mean = red[0] * (1.0f / IN);
    __syncthreads();

    float a = 0.0f;
    for (int k = tid; k < IN; k += 256) {
        float t = fabsf(row[k] - mean);
        a += (t > 1.0f) ? 1.0f : t;
    }
    red[tid] = a;
    __syncthreads();
    for (int off = 128; off > 0; off >>= 1) {
        if (tid < off) red[tid] += red[tid + off];
        __syncthreads();
    }
    if (tid == 0) scale[o] = red[0] * (1.0f / IN);

    const int tn = o >> 4, n = o & 15;
    for (int k = tid; k < IN; k += 256) {
        const float t  = row[k] - mean;
        const int tk   = k >> 7, kk = k & 127;
        const int g    = kk >> 5, rem = kk & 31;
        const int h    = rem >> 4, w16 = rem & 15;
        const int r    = w16 >> 2, j = w16 & 3;
        const int v    = g * 4 + r;
        const int lane = h * 16 + n;
        const size_t addr = (size_t)(tn * KT + tk) * 2048 + lane * 64 + v * 4 + j;
        pb[addr] = (unsigned char)sign_fp8(t);
    }
}

__global__ __launch_bounds__(256) void xnor_gemm_kernel(
    const unsigned char* __restrict__ pa, const unsigned char* __restrict__ pb,
    const float* __restrict__ scale, const float* __restrict__ bias,
    float* __restrict__ out) {
    const int lane = threadIdx.x & 31;
    const int wave = threadIdx.x >> 5;
    const int wm   = wave & 3;
    const int wn   = wave >> 2;
    const int mt0  = blockIdx.y * 8 + wm * 2;
    const int nt0  = blockIdx.x * 8 + wn * 4;

    v8f acc[2][4] = {};

    const unsigned char* aBase[2];
    const unsigned char* bBase[4];
#pragma unroll
    for (int i = 0; i < 2; ++i)
        aBase[i] = pa + (size_t)(mt0 + i) * KT * 2048 + lane * 64;
#pragma unroll
    for (int j = 0; j < 4; ++j)
        bBase[j] = pb + (size_t)(nt0 + j) * KT * 2048 + lane * 64;

    for (int tk = 0; tk < KT; ++tk) {
        const size_t koff = (size_t)tk * 2048;
        v16i a[2], b[4];
#pragma unroll
        for (int i = 0; i < 2; ++i) a[i] = *(const v16i*)(aBase[i] + koff);
#pragma unroll
        for (int j = 0; j < 4; ++j) b[j] = *(const v16i*)(bBase[j] + koff);
#pragma unroll
        for (int i = 0; i < 2; ++i)
#pragma unroll
            for (int j = 0; j < 4; ++j)
                acc[i][j] = __builtin_amdgcn_wmma_f32_16x16x128_fp8_fp8(
                    a[i], b[j], (short)0, acc[i][j], false, false);
    }

#pragma unroll
    for (int j = 0; j < 4; ++j) {
        const int o  = (nt0 + j) * 16 + (lane & 15);
        const float sc = scale[o];
        const float bi = bias[o];
#pragma unroll
        for (int i = 0; i < 2; ++i) {
            const int mbase = (mt0 + i) * 16 + (lane >> 4) * 8;
#pragma unroll
            for (int r = 0; r < 8; ++r)
                out[(size_t)(mbase + r) * OUT + o] = acc[i][j][r] * sc + bi;
        }
    }
}

extern "C" void kernel_launch(void* const* d_in, const int* in_sizes, int n_in,
                              void* d_out, int out_size, void* d_ws, size_t ws_size,
                              hipStream_t stream) {
    const float* x     = (const float*)d_in[0];
    const float* wfull = (const float*)d_in[1];
    const float* bias  = (const float*)d_in[2];
    float* out = (float*)d_out;

    char* ws = (char*)d_ws;
    unsigned*      pa    = (unsigned*)ws;
    unsigned char* pb    = (unsigned char*)(ws + (size_t)BDIM * IN);
    float*         scale = (float*)(ws + (size_t)BDIM * IN + (size_t)OUT * IN);

    const int packDwords = (BDIM / 4) * IN;
    pack_x_kernel<<<packDwords / 256, 256, 0, stream>>>(x, pa);
    pack_w_kernel<<<OUT, 256, 0, stream>>>(wfull, pb, scale);

    dim3 grid(OUT / 128, BDIM / 128);
    xnor_gemm_kernel<<<grid, 256, 0, stream>>>((const unsigned char*)pa, pb, scale,
                                               bias, out);
}
